// pointnet_fp_module_14482629722289
// MI455X (gfx1250) — matrix operand order repaired and checked by reading, not compiled, not run on hardware
//
#include <hip/hip_runtime.h>

typedef __attribute__((ext_vector_type(16))) __bf16 v16bf;
typedef __attribute__((ext_vector_type(8)))  __bf16 v8bf;
typedef __attribute__((ext_vector_type(8)))  float  v8f;
typedef __attribute__((ext_vector_type(4)))  float  v4f;

constexpr int B_  = 8;
constexpr int N2_ = 8192;
constexpr int N1_ = 2048;
constexpr int C1_ = 256;
constexpr int C2_ = 128;
constexpr int K1_ = C1_ + C2_;   // 384
constexpr int M_  = 256;         // output channels (both layers)
constexpr int NT_ = B_ * N2_;    // 65536 total points
constexpr int NBLK_ = NT_ / 128; // 512 GEMM blocks along m
constexpr float EPS_BN = 1e-3f;
constexpr float EPS_D  = 1e-10f;

// ---------------- workspace layout (bytes) ----------------
constexpr size_t OFF_IDX = 0;                                   // int  [NT][3]
constexpr size_t OFF_WGT = OFF_IDX + (size_t)NT_ * 3 * 4;       // f32  [NT][3]
constexpr size_t OFF_W1H = OFF_WGT + (size_t)NT_ * 3 * 4;       // bf16 [256][384]
constexpr size_t OFF_W1L = OFF_W1H + (size_t)M_ * K1_ * 2;
constexpr size_t OFF_W2H = OFF_W1L + (size_t)M_ * K1_ * 2;      // bf16 [256][256]
constexpr size_t OFF_W2L = OFF_W2H + (size_t)M_ * M_ * 2;
constexpr size_t OFF_CH  = OFF_W2L + (size_t)M_ * M_ * 2;       // f32 [256][2] (a,b)
constexpr size_t OFF_PS  = OFF_CH  + 4096;                      // f32 [256][NBLK] partial sums
constexpr size_t OFF_PQ  = OFF_PS  + (size_t)M_ * NBLK_ * 4;    // f32 [256][NBLK] partial sq
constexpr size_t OFF_XH  = OFF_PQ  + (size_t)M_ * NBLK_ * 4;    // bf16 [NT][384]
constexpr size_t OFF_XL  = OFF_XH  + (size_t)NT_ * K1_ * 2;
constexpr size_t OFF_YH  = OFF_XL  + (size_t)NT_ * K1_ * 2;     // bf16 [NT][256]
constexpr size_t OFF_YL  = OFF_YH  + (size_t)NT_ * M_ * 2;

// ---------------- helpers ----------------
__device__ __forceinline__ v16bf comb8(v8bf lo, v8bf hi) {
  v16bf r;
#pragma unroll
  for (int j = 0; j < 8; ++j) { r[j] = lo[j]; r[j + 8] = hi[j]; }
  return r;
}

// ---------------- kernel 0: f32 -> (hi, lo) bf16 split ----------------
__global__ void __launch_bounds__(256)
split_f32_kernel(const float* __restrict__ in, __bf16* __restrict__ oh,
                 __bf16* __restrict__ ol, int n) {
  int i = blockIdx.x * 256 + threadIdx.x;
  if (i < n) {
    float v = in[i];
    __bf16 h = (__bf16)v;
    oh[i] = h;
    ol[i] = (__bf16)(v - (float)h);
  }
}

// ---------------- kernel 1: three_nn + inverse-sq-dist weights ----------------
__global__ void __launch_bounds__(256)
three_nn_kernel(const float* __restrict__ xyz2, const float* __restrict__ xyz1,
                int* __restrict__ idx, float* __restrict__ wgt) {
  const int b = blockIdx.y;
  const int n = blockIdx.x * 256 + threadIdx.x;
  const float px = xyz2[((size_t)b * 3 + 0) * N2_ + n];
  const float py = xyz2[((size_t)b * 3 + 1) * N2_ + n];
  const float pz = xyz2[((size_t)b * 3 + 2) * N2_ + n];
  __shared__ float sx[256], sy[256], sz[256];
  float d0 = 3.0e38f, d1 = 3.0e38f, d2 = 3.0e38f;
  int i0 = 0, i1 = 0, i2 = 0;
  const float* x1 = xyz1 + (size_t)b * 3 * N1_;
  for (int j0 = 0; j0 < N1_; j0 += 256) {
    __syncthreads();
    sx[threadIdx.x] = x1[j0 + threadIdx.x];
    sy[threadIdx.x] = x1[N1_ + j0 + threadIdx.x];
    sz[threadIdx.x] = x1[2 * N1_ + j0 + threadIdx.x];
    __syncthreads();
#pragma unroll 4
    for (int jj = 0; jj < 256; ++jj) {
      float dx = px - sx[jj], dy = py - sy[jj], dz = pz - sz[jj];
      float d = dx * dx + dy * dy + dz * dz;
      int j = j0 + jj;
      if (d < d2) {
        if (d < d1) {
          d2 = d1; i2 = i1;
          if (d < d0) { d1 = d0; i1 = i0; d0 = d; i0 = j; }
          else        { d1 = d;  i1 = j; }
        } else { d2 = d; i2 = j; }
      }
    }
  }
  float v0 = 1.0f / fmaxf(d0, EPS_D);
  float v1 = 1.0f / fmaxf(d1, EPS_D);
  float v2 = 1.0f / fmaxf(d2, EPS_D);
  float inv = 1.0f / (v0 + v1 + v2);
  size_t o = ((size_t)b * N2_ + n) * 3;
  idx[o + 0] = i0; idx[o + 1] = i1; idx[o + 2] = i2;
  wgt[o + 0] = v0 * inv; wgt[o + 1] = v1 * inv; wgt[o + 2] = v2 * inv;
}

// ---------------- kernel 2: interpolate + concat -> X[m][k] bf16 hi/lo ----------------
__global__ void __launch_bounds__(256)
interp_concat_kernel(const float* __restrict__ feat1, const float* __restrict__ feat2,
                     const int* __restrict__ idx, const float* __restrict__ wgt,
                     __bf16* __restrict__ Xh, __bf16* __restrict__ Xl) {
  const int m = blockIdx.x * 256 + threadIdx.x;
  const int b = m >> 13, n = m & 8191;
  size_t o = (size_t)m * 3;
  const int i0 = idx[o], i1 = idx[o + 1], i2 = idx[o + 2];
  const float w0 = wgt[o], w1 = wgt[o + 1], w2 = wgt[o + 2];
  const float* f1 = feat1 + (size_t)b * C1_ * N1_;
  __bf16* xh = Xh + (size_t)m * K1_;
  __bf16* xl = Xl + (size_t)m * K1_;
  for (int cg = 0; cg < C1_; cg += 8) {
    v8bf h, l;
#pragma unroll
    for (int j = 0; j < 8; ++j) {
      const float* row = f1 + (size_t)(cg + j) * N1_;
      float v = w0 * row[i0] + w1 * row[i1] + w2 * row[i2];
      __bf16 hb = (__bf16)v;
      h[j] = hb; l[j] = (__bf16)(v - (float)hb);
    }
    *(v8bf*)(xh + cg) = h;
    *(v8bf*)(xl + cg) = l;
  }
  const float* f2 = feat2 + (size_t)b * C2_ * N2_ + n;
  for (int cg = 0; cg < C2_; cg += 8) {
    v8bf h, l;
#pragma unroll
    for (int j = 0; j < 8; ++j) {
      float v = f2[(size_t)(cg + j) * N2_];
      __bf16 hb = (__bf16)v;
      h[j] = hb; l[j] = (__bf16)(v - (float)hb);
    }
    *(v8bf*)(xh + C1_ + cg) = h;
    *(v8bf*)(xl + C1_ + cg) = l;
  }
}

// ---------------- WMMA GEMM + deterministic BN partial stats ----------------
// Y[o][m] = sum_k W[o][k] * X[k][m], split precision: Ah*Bh + Ah*Bl + Al*Bh.
// Block = 8 waves, block tile 128(o) x 128(m); wave tile 32 x 64.
template <int K, bool TO_OUT>
__global__ void __launch_bounds__(256)
gemm_bn_kernel(const __bf16* __restrict__ Xh, const __bf16* __restrict__ Xl,
               const __bf16* __restrict__ Wh, const __bf16* __restrict__ Wl,
               __bf16* __restrict__ Yh, __bf16* __restrict__ Yl,
               float* __restrict__ outF,
               float* __restrict__ partS, float* __restrict__ partQ) {
  const int lane = threadIdx.x & 31;
  const int wave = threadIdx.x >> 5;
  const int wy = wave >> 1, wx = wave & 1;     // 4 wave-rows x 2 wave-cols
  const int l15 = lane & 15, half = lane >> 4;
  const int o_wave = blockIdx.y * 128 + wy * 32;
  const int m_wave = blockIdx.x * 128 + wx * 64;

  const v8f vzero = {0.f, 0.f, 0.f, 0.f, 0.f, 0.f, 0.f, 0.f};
  v8f acc[2][4];
#pragma unroll
  for (int a = 0; a < 2; ++a)
#pragma unroll
    for (int c = 0; c < 4; ++c) acc[a][c] = vzero;

  // A: lane l15 = row o, kb = half*8 with interleave [kb..kb+7][16+kb..16+kb+7]
  const __bf16* wh0 = Wh + (size_t)(o_wave + l15) * K + half * 8;
  const __bf16* wl0 = Wl + (size_t)(o_wave + l15) * K + half * 8;
  // B: lane l15 = col m, kb = half*8 with interleave [kb..kb+7][16+kb..16+kb+7]
  const __bf16* xh0 = Xh + (size_t)(m_wave + l15) * K + half * 8;
  const __bf16* xl0 = Xl + (size_t)(m_wave + l15) * K + half * 8;

  for (int k0 = 0; k0 < K; k0 += 32) {
    v16bf ah[2], al[2], bh[4], bl[4];
#pragma unroll
    for (int mt = 0; mt < 2; ++mt) {
      const __bf16* p = wh0 + (size_t)mt * 16 * K + k0;
      ah[mt] = comb8(*(const v8bf*)p, *(const v8bf*)(p + 16));
      const __bf16* q = wl0 + (size_t)mt * 16 * K + k0;
      al[mt] = comb8(*(const v8bf*)q, *(const v8bf*)(q + 16));
    }
#pragma unroll
    for (int nt = 0; nt < 4; ++nt) {
      const __bf16* pb = xh0 + (size_t)nt * 16 * K + k0;
      bh[nt] = comb8(*(const v8bf*)pb, *(const v8bf*)(pb + 16));
      const __bf16* qb = xl0 + (size_t)nt * 16 * K + k0;
      bl[nt] = comb8(*(const v8bf*)qb, *(const v8bf*)(qb + 16));
    }
#pragma unroll
    for (int mt = 0; mt < 2; ++mt)
#pragma unroll
      for (int nt = 0; nt < 4; ++nt) {
        acc[mt][nt] = __builtin_amdgcn_wmma_f32_16x16x32_bf16(
            false, ah[mt], false, bh[nt], (short)0, acc[mt][nt], false, false);
        acc[mt][nt] = __builtin_amdgcn_wmma_f32_16x16x32_bf16(
            false, ah[mt], false, bl[nt], (short)0, acc[mt][nt], false, false);
        acc[mt][nt] = __builtin_amdgcn_wmma_f32_16x16x32_bf16(
            false, al[mt], false, bh[nt], (short)0, acc[mt][nt], false, false);
      }
  }

  // deterministic BN partial stats: half-wave shuffle reduce (channel fixed per half)
  __shared__ float ls[128][2];
  __shared__ float lq[128][2];
  float sv[2][8], qv[2][8];
#pragma unroll
  for (int mt = 0; mt < 2; ++mt)
#pragma unroll
    for (int v = 0; v < 8; ++v) {
      float s = 0.f, q = 0.f;
#pragma unroll
      for (int nt = 0; nt < 4; ++nt) { float y = acc[mt][nt][v]; s += y; q += y * y; }
#pragma unroll
      for (int off = 8; off >= 1; off >>= 1) {
        s += __shfl_xor(s, off, 32);
        q += __shfl_xor(q, off, 32);
      }
      sv[mt][v] = s; qv[mt][v] = q;
    }
  if (l15 == 0) {
#pragma unroll
    for (int mt = 0; mt < 2; ++mt)
#pragma unroll
      for (int v = 0; v < 8; ++v) {
        int ch = wy * 32 + mt * 16 + half * 8 + v;
        ls[ch][wx] = sv[mt][v];
        lq[ch][wx] = qv[mt][v];
      }
  }
  __syncthreads();
  if (threadIdx.x < 128) {
    int t = threadIdx.x;
    float S = ls[t][0] + ls[t][1];
    float Q = lq[t][0] + lq[t][1];
    int ch = blockIdx.y * 128 + t;
    partS[(size_t)ch * gridDim.x + blockIdx.x] = S;
    partQ[(size_t)ch * gridDim.x + blockIdx.x] = Q;
  }

  // store results
#pragma unroll
  for (int mt = 0; mt < 2; ++mt) {
    const int cb = o_wave + mt * 16 + half * 8;
#pragma unroll
    for (int nt = 0; nt < 4; ++nt) {
      const int m = m_wave + nt * 16 + l15;
      if constexpr (!TO_OUT) {
        v8bf h, l;
#pragma unroll
        for (int v = 0; v < 8; ++v) {
          float f = acc[mt][nt][v];
          __bf16 hb = (__bf16)f;
          h[v] = hb; l[v] = (__bf16)(f - (float)hb);
        }
        *(v8bf*)(Yh + (size_t)m * M_ + cb) = h;
        *(v8bf*)(Yl + (size_t)m * M_ + cb) = l;
      } else {
        const int bb = m >> 13, n = m & 8191;
        float* op = outF + (size_t)bb * (M_ * N2_) + (size_t)cb * N2_ + n;
#pragma unroll
        for (int v = 0; v < 8; ++v) op[(size_t)v * N2_] = acc[mt][nt][v];
      }
    }
  }
}

// ---------------- BN finalize: per-channel (a,b) from partials (fixed order) ----------------
__global__ void __launch_bounds__(256)
bn_params_kernel(const float* __restrict__ partS, const float* __restrict__ partQ,
                 const float* __restrict__ gamma, const float* __restrict__ beta,
                 float* __restrict__ chpar, int nblk) {
  int c = threadIdx.x;
  float S = 0.f, Q = 0.f;
  for (int i = 0; i < nblk; ++i) {
    S += partS[(size_t)c * nblk + i];
    Q += partQ[(size_t)c * nblk + i];
  }
  const float invN = 1.0f / (float)NT_;
  float mean = S * invN;
  float var = Q * invN - mean * mean;
  float r = rsqrtf(var + EPS_BN);
  float a = gamma[c] * r;
  chpar[2 * c]     = a;
  chpar[2 * c + 1] = beta[c] - mean * a;
}

// ---------------- BN+ReLU apply on bf16-pair planes (in place, re-split) ----------------
__global__ void __launch_bounds__(256)
bn_relu_split_kernel(__bf16* __restrict__ Yh, __bf16* __restrict__ Yl,
                     const float* __restrict__ chpar) {
  size_t e0 = ((size_t)blockIdx.x * 256 + threadIdx.x) * 8;
  int c = (int)(e0 & 255);
  v8bf h = *(v8bf*)(Yh + e0);
  v8bf l = *(v8bf*)(Yl + e0);
  v8bf ho, lo;
#pragma unroll
  for (int j = 0; j < 8; ++j) {
    float x = (float)h[j] + (float)l[j];
    float a = chpar[2 * (c + j)], b = chpar[2 * (c + j) + 1];
    float y = fmaxf(a * x + b, 0.0f);
    __bf16 hb = (__bf16)y;
    ho[j] = hb; lo[j] = (__bf16)(y - (float)hb);
  }
  *(v8bf*)(Yh + e0) = ho;
  *(v8bf*)(Yl + e0) = lo;
}

// ---------------- BN+ReLU apply on f32 output (in place) ----------------
__global__ void __launch_bounds__(256)
bn_relu_out_kernel(float* __restrict__ out, const float* __restrict__ chpar) {
  size_t e0 = ((size_t)blockIdx.x * 256 + threadIdx.x) * 4;
  int c = (int)((e0 >> 13) & 255);
  float a = chpar[2 * c], b = chpar[2 * c + 1];
  v4f x = *(v4f*)(out + e0);
  v4f y;
#pragma unroll
  for (int j = 0; j < 4; ++j) y[j] = fmaxf(a * x[j] + b, 0.0f);
  *(v4f*)(out + e0) = y;
}

// ---------------- launcher ----------------
extern "C" void kernel_launch(void* const* d_in, const int* in_sizes, int n_in,
                              void* d_out, int out_size, void* d_ws, size_t ws_size,
                              hipStream_t stream) {
  (void)in_sizes; (void)n_in; (void)out_size; (void)ws_size;
  const float* xyz2  = (const float*)d_in[0];
  const float* xyz1  = (const float*)d_in[1];
  const float* feat2 = (const float*)d_in[2];
  const float* feat1 = (const float*)d_in[3];
  const float* W1    = (const float*)d_in[4];
  // d_in[5] = b1: conv bias cancels exactly under training-mode BN -> skipped
  const float* g1    = (const float*)d_in[6];
  const float* be1   = (const float*)d_in[7];
  const float* W2    = (const float*)d_in[8];
  // d_in[9] = b2: skipped (cancels)
  const float* g2    = (const float*)d_in[10];
  const float* be2   = (const float*)d_in[11];
  float* out = (float*)d_out;

  char* ws = (char*)d_ws;
  int*    idx   = (int*)(ws + OFF_IDX);
  float*  wgt   = (float*)(ws + OFF_WGT);
  __bf16* W1h   = (__bf16*)(ws + OFF_W1H);
  __bf16* W1l   = (__bf16*)(ws + OFF_W1L);
  __bf16* W2h   = (__bf16*)(ws + OFF_W2H);
  __bf16* W2l   = (__bf16*)(ws + OFF_W2L);
  float*  chpar = (float*)(ws + OFF_CH);
  float*  partS = (float*)(ws + OFF_PS);
  float*  partQ = (float*)(ws + OFF_PQ);
  __bf16* Xh    = (__bf16*)(ws + OFF_XH);
  __bf16* Xl    = (__bf16*)(ws + OFF_XL);
  __bf16* Yh    = (__bf16*)(ws + OFF_YH);
  __bf16* Yl    = (__bf16*)(ws + OFF_YL);

  // 0. weight split to bf16 hi/lo
  split_f32_kernel<<<dim3((M_ * K1_ + 255) / 256), 256, 0, stream>>>(W1, W1h, W1l, M_ * K1_);
  split_f32_kernel<<<dim3((M_ * M_ + 255) / 256), 256, 0, stream>>>(W2, W2h, W2l, M_ * M_);
  // 1. three_nn + weights
  three_nn_kernel<<<dim3(N2_ / 256, B_), 256, 0, stream>>>(xyz2, xyz1, idx, wgt);
  // 2. interpolate + concat into X[m][k] hi/lo
  interp_concat_kernel<<<dim3(NT_ / 256), 256, 0, stream>>>(feat1, feat2, idx, wgt, Xh, Xl);
  // 3. layer-1 GEMM (K=384) -> Y hi/lo + BN partials
  gemm_bn_kernel<K1_, false><<<dim3(NT_ / 128, M_ / 128), 256, 0, stream>>>(
      Xh, Xl, W1h, W1l, Yh, Yl, nullptr, partS, partQ);
  // 4. BN params layer 1
  bn_params_kernel<<<1, 256, 0, stream>>>(partS, partQ, g1, be1, chpar, NBLK_);
  // 5. BN+ReLU apply on Y (in place)
  bn_relu_split_kernel<<<dim3((NT_ * M_) / (256 * 8)), 256, 0, stream>>>(Yh, Yl, chpar);
  // 6. layer-2 GEMM (K=256) -> d_out (pre-BN) + BN partials
  gemm_bn_kernel<M_, true><<<dim3(NT_ / 128, M_ / 128), 256, 0, stream>>>(
      Yh, Yl, W2h, W2l, nullptr, nullptr, out, partS, partQ);
  // 7. BN params layer 2
  bn_params_kernel<<<1, 256, 0, stream>>>(partS, partQ, g2, be2, chpar, NBLK_);
  // 8. BN+ReLU apply on d_out (in place)
  bn_relu_out_kernel<<<dim3((NT_ * M_) / (256 * 4)), 256, 0, stream>>>(out, chpar);
}
